// MultiheadAttention_15659450761322
// MI455X (gfx1250) — matrix operand order repaired and checked by reading, not compiled, not run on hardware
//
#include <hip/hip_runtime.h>
#include <hip/hip_bf16.h>

#define BATCH 4
#define SEQ   2048
#define EMB   1024
#define NHEAD 16
#define DHEAD 64
#define MROWS (BATCH * SEQ)   // 8192
#define NKT   (SEQ / 32)      // 64 key tiles

typedef __attribute__((ext_vector_type(16))) __bf16       bf16x16;
typedef __attribute__((ext_vector_type(8)))  float        f32x8;
typedef __attribute__((ext_vector_type(4)))  unsigned int u32x4;

// ---------------------------------------------------------------- WMMA helper
__device__ inline f32x8 wmma_bf16(bf16x16 a, bf16x16 b, f32x8 c) {
  return __builtin_amdgcn_wmma_f32_16x16x32_bf16(
      /*neg_a=*/false, a, /*neg_b=*/false, b,
      /*c_mod=*/(short)0, c, /*reuse_a=*/false, /*reuse_b=*/false);
}

// B-fragment (32x16 K-major): lane holds 16 *contiguous* K values -> 2x b128
__device__ inline bf16x16 ldfrag(const __bf16* p) {
  union { u32x4 u[2]; bf16x16 v; } t;
  t.u[0] = *(const u32x4*)(p);
  t.u[1] = *(const u32x4*)(p + 8);
  return t.v;
}

// A-fragment (16x32): lane holds 16 contiguous K values [k..k+15] -> 2x b128
__device__ inline bf16x16 ldfrag_a(const __bf16* p) {
  union { u32x4 u[2]; bf16x16 v; } t;
  t.u[0] = *(const u32x4*)(p);
  t.u[1] = *(const u32x4*)(p + 8);
  return t.v;
}

// A-fragment built from f32 memory (convert on the fly)
__device__ inline bf16x16 ldfrag_a_f32(const float* p) {
  bf16x16 r;
#pragma unroll
  for (int i = 0; i < 8; ++i) r[i] = (__bf16)p[i];
#pragma unroll
  for (int i = 0; i < 8; ++i) r[8 + i] = (__bf16)p[8 + i];
  return r;
}

// ---------------------------------------------- CDNA5 async copy to LDS (GV)
// LDS byte address comes from the vdst VGPR; generic __shared__ ptr low 32 bits == LDS off
__device__ inline void async_ld_b128(unsigned lds_byte_off, const void* g) {
  asm volatile("global_load_async_to_lds_b128 %0, %1, off"
               :: "v"(lds_byte_off), "v"((unsigned long long)(uintptr_t)g)
               : "memory");
}
#define ASYNC_WAIT(imm) asm volatile("s_wait_asynccnt " imm ::: "memory")

// ------------------------------------------------- weight transpose + convert
// W[K,N] f32 -> Wt[N,K] bf16 via LDS tile (coalesced both sides)
__global__ __launch_bounds__(256) void wtrans_kernel(const float* __restrict__ W,
                                                     __bf16* __restrict__ Wt) {
  __shared__ float tile[32][33];
  const int bx = blockIdx.x * 32, by = blockIdx.y * 32;
#pragma unroll
  for (int j = 0; j < 32; j += 8)
    tile[threadIdx.y + j][threadIdx.x] =
        W[(size_t)(by + threadIdx.y + j) * EMB + bx + threadIdx.x];
  __syncthreads();
#pragma unroll
  for (int j = 0; j < 32; j += 8)
    Wt[(size_t)(bx + threadIdx.y + j) * EMB + by + threadIdx.x] =
        (__bf16)tile[threadIdx.x][threadIdx.y + j];
}

// ------------------------------------------------------------ projection GEMM
// C[M,N] = X[M,K]*Wt[N,K]^T + bias ; wave computes a 32x64 tile (8 C tiles).
// MODE 0: write bf16 [B,H,S,D] (Q,K)   MODE 2: write bf16 [B,H,D,S] (V^T)
template <int MODE>
__global__ __launch_bounds__(256) void proj_gemm_kernel(
    const float* __restrict__ X, const __bf16* __restrict__ Wt,
    const float* __restrict__ bias, __bf16* __restrict__ out) {
  const int lane = threadIdx.x & 31;
  const int wv   = threadIdx.x >> 5;
  const int l16  = lane & 15;
  const int half = lane >> 4;
  const int row0 = blockIdx.x * 32;
  const int col0 = (blockIdx.y * 8 + wv) * 64;

  f32x8 acc[2][4] = {};
  const float*  Arow = X  + (size_t)(row0 + l16) * EMB + half * 16;
  const __bf16* Brow = Wt + (size_t)(col0 + l16) * EMB + half * 16;

  for (int kk = 0; kk < EMB; kk += 32) {
    __builtin_prefetch(Brow + kk + 128, 0, 1);
    __builtin_prefetch(Arow + kk + 128, 0, 1);
    const float* a0p = Arow + kk;
    bf16x16 a0 = ldfrag_a_f32(a0p);
    bf16x16 a1 = ldfrag_a_f32(a0p + (size_t)16 * EMB);
    bf16x16 b[4];
#pragma unroll
    for (int t = 0; t < 4; ++t) b[t] = ldfrag(Brow + kk + (size_t)(t * 16) * EMB);
#pragma unroll
    for (int t = 0; t < 4; ++t) {
      acc[0][t] = wmma_bf16(a0, b[t], acc[0][t]);
      acc[1][t] = wmma_bf16(a1, b[t], acc[1][t]);
    }
  }

#pragma unroll
  for (int tm = 0; tm < 2; ++tm) {
#pragma unroll
    for (int tn = 0; tn < 4; ++tn) {
      const int c  = col0 + tn * 16 + l16;
      const float bb = bias[c];
      const int hh = c >> 6, dd = c & (DHEAD - 1);
#pragma unroll
      for (int r = 0; r < 8; ++r) {
        const int m = row0 + tm * 16 + r + 8 * half;
        const int bi = m >> 11, si = m & (SEQ - 1);
        const float v = acc[tm][tn][r] + bb;
        size_t idx;
        if (MODE == 0)
          idx = (((size_t)(bi * NHEAD + hh) * SEQ) + si) * DHEAD + dd;
        else
          idx = (((size_t)(bi * NHEAD + hh) * DHEAD) + dd) * SEQ + si;
        out[idx] = (__bf16)v;
      }
    }
  }
}

// ------------------------------------------------------------- output GEMM
// out[M,N] f32 = A[M,K] bf16 * Wt[N,K]^T + bias ; 32x64 tile per wave
__global__ __launch_bounds__(256) void out_gemm_kernel(
    const __bf16* __restrict__ A, const __bf16* __restrict__ Wt,
    const float* __restrict__ bias, float* __restrict__ out) {
  const int lane = threadIdx.x & 31;
  const int wv   = threadIdx.x >> 5;
  const int l16  = lane & 15;
  const int half = lane >> 4;
  const int row0 = blockIdx.x * 32;
  const int col0 = (blockIdx.y * 8 + wv) * 64;

  f32x8 acc[2][4] = {};
  const __bf16* Arow = A  + (size_t)(row0 + l16) * EMB + half * 16;
  const __bf16* Brow = Wt + (size_t)(col0 + l16) * EMB + half * 16;

  for (int kk = 0; kk < EMB; kk += 32) {
    __builtin_prefetch(Brow + kk + 128, 0, 1);
    __builtin_prefetch(Arow + kk + 128, 0, 1);
    bf16x16 a0 = ldfrag_a(Arow + kk);
    bf16x16 a1 = ldfrag_a(Arow + kk + (size_t)16 * EMB);
    bf16x16 b[4];
#pragma unroll
    for (int t = 0; t < 4; ++t) b[t] = ldfrag(Brow + kk + (size_t)(t * 16) * EMB);
#pragma unroll
    for (int t = 0; t < 4; ++t) {
      acc[0][t] = wmma_bf16(a0, b[t], acc[0][t]);
      acc[1][t] = wmma_bf16(a1, b[t], acc[1][t]);
    }
  }

#pragma unroll
  for (int tm = 0; tm < 2; ++tm) {
#pragma unroll
    for (int tn = 0; tn < 4; ++tn) {
      const int c = col0 + tn * 16 + l16;
      const float bb = bias[c];
#pragma unroll
      for (int r = 0; r < 8; ++r) {
        const int m = row0 + tm * 16 + r + 8 * half;
        out[(size_t)m * EMB + c] = acc[tm][tn][r] + bb;
      }
    }
  }
}

// -------------------------------------------------------- fused attention
// Block = 8 waves sharing one (b,h), 128 consecutive queries (16 per wave).
// K/V 32-key tiles are double-buffered in LDS via async-to-LDS DMA and shared
// by all waves. Online softmax keeps stats in the transposed (row-per-lane)
// layout so row reductions are per-lane + 2 shuffles.
// Q,K: [B,H,S,D] bf16 ; Vt: [B,H,D,S] bf16 ; O: [B,S,E] bf16
__global__ __launch_bounds__(256) void flash_attn_kernel(
    const __bf16* __restrict__ Q, const __bf16* __restrict__ Km,
    const __bf16* __restrict__ Vt, __bf16* __restrict__ O) {
  __shared__ __attribute__((aligned(16))) __bf16 kbuf[2][32 * DHEAD]; // 2x4KB
  __shared__ __attribute__((aligned(16))) __bf16 vbuf[2][DHEAD * 32]; // 2x4KB
  __shared__ __attribute__((aligned(16))) __bf16 ptile[8][16 * 32];   // 8x1KB

  const int tid  = threadIdx.x;
  const int lane = tid & 31;
  const int wv   = tid >> 5;
  const int l16  = lane & 15;
  const int half = lane >> 4;

  const int bh = blockIdx.x >> 4;               // b*H + h
  const int q0 = (blockIdx.x & 15) * 128 + wv * 16;
  const int h  = bh & (NHEAD - 1);
  const int b  = bh >> 4;

  const __bf16* Qb = Q  + ((size_t)bh * SEQ + q0) * DHEAD;
  const __bf16* Kb = Km + (size_t)bh * SEQ * DHEAD;
  const __bf16* Vb = Vt + (size_t)bh * DHEAD * SEQ;
  __bf16* pt = ptile[wv];

  const unsigned koff0 = (unsigned)(uintptr_t)(void*)&kbuf[0][0];
  const unsigned voff0 = (unsigned)(uintptr_t)(void*)&vbuf[0][0];

  // Q A-fragments for the two 32-wide k-steps across D=64; kept resident
  bf16x16 qf0 = ldfrag_a(Qb + l16 * DHEAD + half * 16);
  bf16x16 qf1 = ldfrag_a(Qb + l16 * DHEAD + 32 + half * 16);

  f32x8 acc[4] = {};            // O accumulators: 4 d-tiles of 16
  float mt = -3.0e38f, lt = 0.f;  // stats for row l16 (transposed layout)

  // issue one 32-key K tile (4KB contiguous) + V tile (64 rows x 64B) per step
  auto issue = [&](int kt) {
    const int buf = kt & 1;
    const char* kg = (const char*)(Kb + (size_t)kt * 32 * DHEAD) + tid * 16;
    async_ld_b128(koff0 + buf * 4096u + tid * 16, kg);
    const int vr = tid >> 2, vc = (tid & 3) * 16;     // row 0..63, 16B chunk
    const char* vg = (const char*)(Vb + (size_t)vr * SEQ + kt * 32) + vc;
    async_ld_b128(voff0 + buf * 4096u + vr * 64 + vc, vg);
  };

  issue(0);
  for (int kt = 0; kt < NKT; ++kt) {
    if (kt + 1 < NKT) { issue(kt + 1); ASYNC_WAIT("0x2"); }
    else              { ASYNC_WAIT("0x0"); }
    __syncthreads();

    const __bf16* kb = kbuf[kt & 1];
    const __bf16* vb = vbuf[kt & 1];

    // logits: 16 q x 32 keys via 4 WMMAs (K tile from LDS)
    bf16x16 k00 = ldfrag(kb + l16 * DHEAD + half * 16);
    bf16x16 k01 = ldfrag(kb + l16 * DHEAD + 32 + half * 16);
    bf16x16 k10 = ldfrag(kb + (16 + l16) * DHEAD + half * 16);
    bf16x16 k11 = ldfrag(kb + (16 + l16) * DHEAD + 32 + half * 16);

    f32x8 z = {};
    f32x8 c0 = wmma_bf16(qf0, k00, z);
    c0 = wmma_bf16(qf1, k01, c0);
    f32x8 c1 = wmma_bf16(qf0, k10, z);
    c1 = wmma_bf16(qf1, k11, c1);

    // stage scaled logits through per-wave LDS tile: C layout -> A layout
#pragma unroll
    for (int r = 0; r < 8; ++r) {
      pt[(r + 8 * half) * 32 + l16]      = (__bf16)(c0[r] * 0.125f);
      pt[(r + 8 * half) * 32 + 16 + l16] = (__bf16)(c1[r] * 0.125f);
    }
    asm volatile("" ::: "memory");   // same-wave DS ops are HW in-order
    bf16x16 slog = ldfrag_a(pt + l16 * 32 + half * 16);
    asm volatile("" ::: "memory");

    // per-lane row softmax (row = l16; split across lanes l16 and l16+16)
    float s[16];
#pragma unroll
    for (int i = 0; i < 16; ++i) s[i] = (float)slog[i];
    float tmax = s[0];
#pragma unroll
    for (int i = 1; i < 16; ++i) tmax = fmaxf(tmax, s[i]);
    tmax = fmaxf(tmax, __shfl_xor(tmax, 16, 32));
    const float nm = fmaxf(mt, tmax);
    const float sc = __expf(mt - nm);
    mt = nm;
    bf16x16 pf;
    float psum = 0.f;
#pragma unroll
    for (int i = 0; i < 16; ++i) {
      const float p = __expf(s[i] - nm);
      psum += p;
      pf[i] = (__bf16)p;
    }
    psum += __shfl_xor(psum, 16, 32);
    lt = lt * sc + psum;

    // rescale accumulators (C layout rows r + 8*half) with broadcast scales
#pragma unroll
    for (int r = 0; r < 8; ++r) {
      const float scr = __shfl(sc, r + 8 * half, 32);
#pragma unroll
      for (int t = 0; t < 4; ++t) acc[t][r] *= scr;
    }

    // O += P @ V  (V tile from LDS, [64 d][32 k] row-major)
#pragma unroll
    for (int t = 0; t < 4; ++t) {
      bf16x16 vf = ldfrag(vb + (t * 16 + l16) * 32 + half * 16);
      acc[t] = wmma_bf16(pf, vf, acc[t]);
    }
    __syncthreads();   // protect LDS buffers before next overwrite
  }

  // normalize (broadcast l back to C layout) and scatter to O[B,S,E]
#pragma unroll
  for (int r = 0; r < 8; ++r) {
    const float lr = __shfl(lt, r + 8 * half, 32);
    const int qrow = q0 + r + 8 * half;
#pragma unroll
    for (int t = 0; t < 4; ++t) {
      const float v = acc[t][r] / lr;
      O[((size_t)b * SEQ + qrow) * EMB + h * DHEAD + t * 16 + l16] = (__bf16)v;
    }
  }
}

// ------------------------------------------------------------------ launcher
extern "C" void kernel_launch(void* const* d_in, const int* in_sizes, int n_in,
                              void* d_out, int out_size, void* d_ws, size_t ws_size,
                              hipStream_t stream) {
  const float* query = (const float*)d_in[0];
  const float* key_i = (const float*)d_in[1];
  const float* value = (const float*)d_in[2];
  const float* Wq = (const float*)d_in[3];
  const float* bq = (const float*)d_in[4];
  const float* Wk = (const float*)d_in[5];
  const float* bk = (const float*)d_in[6];
  const float* Wv = (const float*)d_in[7];
  const float* bv = (const float*)d_in[8];
  const float* Wo = (const float*)d_in[9];
  const float* bo = (const float*)d_in[10];

  char* ws = (char*)d_ws;
  const size_t WBYTES = (size_t)EMB * EMB * sizeof(__bf16);          // 2 MB each
  const size_t TBYTES = (size_t)BATCH * NHEAD * SEQ * DHEAD * 2;     // 16 MB each
  __bf16* Wq_t = (__bf16*)(ws);
  __bf16* Wk_t = (__bf16*)(ws + WBYTES);
  __bf16* Wv_t = (__bf16*)(ws + 2 * WBYTES);
  __bf16* Wo_t = (__bf16*)(ws + 3 * WBYTES);
  __bf16* Qb   = (__bf16*)(ws + 4 * WBYTES);
  __bf16* Kb   = (__bf16*)(ws + 4 * WBYTES + TBYTES);
  __bf16* Vtb  = (__bf16*)(ws + 4 * WBYTES + 2 * TBYTES);
  __bf16* Ob   = (__bf16*)(ws + 4 * WBYTES + 3 * TBYTES);            // 72 MB total

  dim3 tb(32, 8);
  dim3 tg(EMB / 32, EMB / 32);
  wtrans_kernel<<<tg, tb, 0, stream>>>(Wq, Wq_t);
  wtrans_kernel<<<tg, tb, 0, stream>>>(Wk, Wk_t);
  wtrans_kernel<<<tg, tb, 0, stream>>>(Wv, Wv_t);
  wtrans_kernel<<<tg, tb, 0, stream>>>(Wo, Wo_t);

  dim3 gg(MROWS / 32, EMB / 512);  // 256 x 2 blocks, 8 waves each (32x64/wave)
  proj_gemm_kernel<0><<<gg, 256, 0, stream>>>(query, Wq_t, bq, Qb);
  proj_gemm_kernel<0><<<gg, 256, 0, stream>>>(key_i, Wk_t, bk, Kb);
  proj_gemm_kernel<2><<<gg, 256, 0, stream>>>(value, Wv_t, bv, Vtb);

  // 64 (b,h) x 16 q-blocks of 128 queries
  flash_attn_kernel<<<BATCH * NHEAD * (SEQ / 128), 256, 0, stream>>>(Qb, Kb, Vtb, Ob);

  out_gemm_kernel<<<gg, 256, 0, stream>>>(Ob, Wo_t, bo, (float*)d_out);
}
